// ParallelModel_6545530159462
// MI455X (gfx1250) — matrix operand order repaired and checked by reading, not compiled, not run on hardware
//
#include <hip/hip_runtime.h>
#include <hip/hip_bf16.h>

typedef __attribute__((ext_vector_type(2)))  float    v2f;
typedef __attribute__((ext_vector_type(8)))  float    v8f;
typedef __attribute__((ext_vector_type(16))) _Float16 v16h;

#define RAYS   16384
#define NS     64
#define NPTS   (RAYS * NS)   // 1,048,576 points per pass
#define ZNEAR  2.0f
#define ZSPAN  4.0f          // FAR - NEAR

// ---------------------------------------------------------------------------
// Kernel 1: normalize ray dirs, compute coarse sample points. 1 thread/point.
// ---------------------------------------------------------------------------
__global__ __launch_bounds__(256) void nerf_setup(
    const float* __restrict__ org, const float* __restrict__ dir,
    const float* __restrict__ noise, float* __restrict__ rd, float* __restrict__ pts)
{
    int p   = blockIdx.x * 256 + threadIdx.x;
    int ray = p >> 6, n = p & 63;
    float dx = dir[ray*3+0], dy = dir[ray*3+1], dz = dir[ray*3+2];
    float inv = 1.0f / sqrtf(dx*dx + dy*dy + dz*dz);
    dx *= inv; dy *= inv; dz *= inv;
    if (n == 0) { rd[ray*3+0] = dx; rd[ray*3+1] = dy; rd[ray*3+2] = dz; }
    float z = ZNEAR + ZSPAN * (((float)n + noise[p]) * (1.0f/64.0f));
    float ox = org[ray*3+0], oy = org[ray*3+1], oz = org[ray*3+2];
    pts[p*3+0] = ox + dx*z;
    pts[p*3+1] = oy + dy*z;
    pts[p*3+2] = oz + dz*z;
}

// ---------------------------------------------------------------------------
// Kernel 2: WMMA MLP.  Each wave handles 16 points.
//   Layer1: A(16x4 f32 pts) x B(4x16 W1 tile) via V_WMMA_F32_16X16X4_F32, x4
//   Layer2: h->f16, A(16x32) x B(32x16 W2 pad) via V_WMMA_F32_16X16X32_F16, x2
// ---------------------------------------------------------------------------
__global__ __launch_bounds__(256) void nerf_mlp(
    const float* __restrict__ pts,
    const float* __restrict__ W1p, const float* __restrict__ b1p,
    const float* __restrict__ W2p, const float* __restrict__ b2p,
    float* __restrict__ sigma, float* __restrict__ rgb)
{
    __shared__ float    s_w1[4][64];    // K-padded W1
    __shared__ float    s_b1[64];
    __shared__ _Float16 s_w2[64][16];   // N-padded W2 (f16)
    __shared__ float    s_b2[4];
    __shared__ float    s_h[8][16][64]; // per-wave activations

    int tid = threadIdx.x;
    { int k = tid >> 6, n = tid & 63; s_w1[k][n] = (k < 3) ? W1p[k*64 + n] : 0.0f; }
    if (tid < 64) s_b1[tid] = b1p[tid];
    for (int i = tid; i < 1024; i += 256) {
        int k = i >> 4, n = i & 15;
        s_w2[k][n] = (n < 4) ? (_Float16)W2p[k*4 + n] : (_Float16)0.0f;
    }
    if (tid < 4) s_b2[tid] = b2p[tid];
    __syncthreads();

    int wave = tid >> 5, lane = tid & 31;
    int M = lane & 15, hi = lane >> 4;
    int p0 = (blockIdx.x * 8 + wave) * 16;

    // ----- layer 1 -----
    int K0 = hi * 2;                    // A f32 16x4 layout: K = 2*(lane/16)+r
    int pt = p0 + M;
    v2f a;
    a.x = (K0     < 3) ? pts[pt*3 + K0    ] : 0.0f;
    a.y = (K0 + 1 < 3) ? pts[pt*3 + K0 + 1] : 0.0f;

    v8f hacc[4];
#pragma unroll
    for (int t = 0; t < 4; ++t) {
        v2f bm;
        bm.x = s_w1[K0    ][t*16 + M];  // B: N = lane%16, K = 2*(lane/16)+r
        bm.y = s_w1[K0 + 1][t*16 + M];
        v8f c = {};
        hacc[t] = __builtin_amdgcn_wmma_f32_16x16x4_f32(
            false, a, false, bm, (short)0, c, false, false);
    }
    // bias + ReLU, stage to LDS in [point][hidden] order
#pragma unroll
    for (int t = 0; t < 4; ++t) {
        int n = t*16 + M;
        float bb = s_b1[n];
#pragma unroll
        for (int r = 0; r < 8; ++r) {
            float v = hacc[t][r] + bb;              // C layout: m = r + 8*hi
            s_h[wave][r + 8*hi][n] = v > 0.0f ? v : 0.0f;
        }
    }
    __syncthreads();

    // ----- layer 2 -----
    v16h a0, a1;
#pragma unroll
    for (int i = 0; i < 16; ++i) {      // A f16 16x32: K = 8*hi+i | 16+8*hi+(i-8)
        int K = (i < 8) ? (hi*8 + i) : (16 + hi*8 + (i - 8));
        a0[i] = (_Float16)s_h[wave][M][K];
        a1[i] = (_Float16)s_h[wave][M][32 + K];
    }
    v16h bb0, bb1;
#pragma unroll
    for (int i = 0; i < 16; ++i) {      // B f16: N = lane%16, K = 8*hi+i | 16+8*hi+(i-8)
        int K = (i < 8) ? (hi*8 + i) : (16 + hi*8 + (i - 8));
        bb0[i] = s_w2[K][M];
        bb1[i] = s_w2[32 + K][M];
    }
    v8f acc = {};
    acc = __builtin_amdgcn_wmma_f32_16x16x32_f16(false, a0, false, bb0, (short)0, acc, false, false);
    acc = __builtin_amdgcn_wmma_f32_16x16x32_f16(false, a1, false, bb1, (short)0, acc, false, false);

    if (M < 4) {
        float bb = s_b2[M];
#pragma unroll
        for (int r = 0; r < 8; ++r) {
            int ptg = p0 + r + 8*hi;
            float v = acc[r] + bb;
            if (M == 0) sigma[ptg] = v;
            else        rgb[ptg*3 + (M-1)] = 1.0f / (1.0f + __expf(-v));
        }
    }
}

// ---------------------------------------------------------------------------
// Kernel 3: per-ray importance sampling. One wave per ray, 8 rays per block.
// ---------------------------------------------------------------------------
__global__ __launch_bounds__(256) void nerf_sample(
    const float* __restrict__ noise, const float* __restrict__ usamp,
    const float* __restrict__ sigma_c, const float* __restrict__ org,
    const float* __restrict__ rd, float* __restrict__ zs, float* __restrict__ pts)
{
    __shared__ float s_z[8][64], s_w[8][64], s_sm[8][64], s_cdf[8][64], s_zm[8][64];
    int wave = threadIdx.x >> 5, lane = threadIdx.x & 31;
    int ray  = blockIdx.x * 8 + wave;

    for (int j = lane; j < 64; j += 32)
        s_z[wave][j] = ZNEAR + ZSPAN * (((float)j + noise[ray*64 + j]) * (1.0f/64.0f));
    __syncthreads();

    // alpha + cumprod (two consecutive samples per lane, shuffle scan)
    int n0 = 2*lane, n1 = 2*lane + 1;
    float z0 = s_z[wave][n0], z1 = s_z[wave][n1];
    float d0 = z1 - z0;
    float d1 = (n1 < 63) ? (s_z[wave][n1+1] - z1) : 1e10f;
    float sg0 = fmaxf(sigma_c[ray*64 + n0], 0.0f);
    float sg1 = fmaxf(sigma_c[ray*64 + n1], 0.0f);
    float al0 = 1.0f - __expf(-sg0*d0);
    float al1 = 1.0f - __expf(-sg1*d1);
    float t0 = 1.0f - al0 + 1e-10f, t1 = 1.0f - al1 + 1e-10f;
    float s = t0 * t1;
    for (int off = 1; off < 32; off <<= 1) { float v = __shfl_up(s, off, 32); if (lane >= off) s *= v; }
    float excl = __shfl_up(s, 1, 32); if (lane == 0) excl = 1.0f;
    s_w[wave][n0] = al0 * excl;
    s_w[wave][n1] = al1 * excl * t0;
    __syncthreads();

    // max-pool(k2,s1,p1) then avg-pool(k2,s1), +0.01  ->  sm[0..63]
    for (int j = lane; j < 64; j += 32) {
        float wm1 = (j >= 1 ) ? s_w[wave][j-1] : -1e30f;
        float wj  = s_w[wave][j];
        float wp1 = (j < 63) ? s_w[wave][j+1] : -1e30f;
        s_sm[wave][j] = 0.5f * (fmaxf(wm1, wj) + fmaxf(wj, wp1)) + 0.01f;
        if (j < 63) s_zm[wave][j] = 0.5f * (s_z[wave][j] + s_z[wave][j+1]);
    }
    __syncthreads();

    // CDF over sm[1..62] (62 values -> cdf[0..62], cdf[0]=0)
    float u0v = 0.0f, u1v = 0.0f;
    if (lane < 31) { u0v = s_sm[wave][2*lane + 1]; u1v = s_sm[wave][2*lane + 2]; }
    float ss = u0v + u1v;
    for (int off = 1; off < 32; off <<= 1) { float v = __shfl_up(ss, off, 32); if (lane >= off) ss += v; }
    float total = __shfl(ss, 31, 32);
    float sexcl = __shfl_up(ss, 1, 32); if (lane == 0) sexcl = 0.0f;
    float invT = 1.0f / total;
    if (lane == 0) s_cdf[wave][0] = 0.0f;
    if (lane < 31) {
        s_cdf[wave][2*lane + 1] = (sexcl + u0v) * invT;
        s_cdf[wave][2*lane + 2] = (sexcl + u0v + u1v) * invT;
    }
    __syncthreads();

    // searchsorted(right) + lerp -> fine z and fine points
    float ox = org[ray*3+0], oy = org[ray*3+1], oz = org[ray*3+2];
    float dx = rd[ray*3+0], dy = rd[ray*3+1], dzv = rd[ray*3+2];
    for (int j = lane; j < 64; j += 32) {
        float u = usamp[ray*64 + j];
        int lo = 0, hii = 63;
        while (lo < hii) { int mid = (lo + hii) >> 1; if (s_cdf[wave][mid] <= u) lo = mid + 1; else hii = mid; }
        int below = lo - 1; below = below < 0 ? 0 : (below > 62 ? 62 : below);
        int above = lo > 62 ? 62 : lo;
        float cb = s_cdf[wave][below], ca = s_cdf[wave][above];
        float bb = s_zm[wave][below],  ba = s_zm[wave][above];
        float den = ca - cb; if (den < 1e-5f) den = 1.0f;
        float zv = bb + ((u - cb) / den) * (ba - bb);
        int p = ray*64 + j;
        zs[p] = zv;
        pts[p*3+0] = ox + dx*zv;
        pts[p*3+1] = oy + dy*zv;
        pts[p*3+2] = oz + dzv*zv;
    }
}

// ---------------------------------------------------------------------------
// Kernel 4: merge-sort (bitonic, 128 in LDS) + final compositing + outputs.
// One wave per ray, 8 rays per block.
// ---------------------------------------------------------------------------
__global__ __launch_bounds__(256) void nerf_final(
    const float* __restrict__ noise, const float* __restrict__ zs,
    const float* __restrict__ sigma_c, const float* __restrict__ rgb_c,
    const float* __restrict__ sigma_f, const float* __restrict__ rgb_f,
    const float* __restrict__ rd, const float* __restrict__ tform,
    float* __restrict__ out)
{
    __shared__ float s_key[8][128];
    __shared__ int   s_idx[8][128];
    int wave = threadIdx.x >> 5, lane = threadIdx.x & 31;
    int ray  = blockIdx.x * 8 + wave;

    for (int j = lane; j < 128; j += 32) {
        float z = (j < 64)
            ? ZNEAR + ZSPAN * (((float)j + noise[ray*64 + j]) * (1.0f/64.0f))
            : zs[ray*64 + (j - 64)];
        s_key[wave][j] = z;
        s_idx[wave][j] = j;
    }
    __syncthreads();

    for (int k = 2; k <= 128; k <<= 1) {
        for (int j = k >> 1; j > 0; j >>= 1) {
            for (int i = lane; i < 128; i += 32) {
                int ixj = i ^ j;
                if (ixj > i) {
                    bool up = ((i & k) == 0);
                    float a = s_key[wave][i], b = s_key[wave][ixj];
                    if (up ? (a > b) : (a < b)) {
                        s_key[wave][i] = b; s_key[wave][ixj] = a;
                        int ti = s_idx[wave][i];
                        s_idx[wave][i] = s_idx[wave][ixj]; s_idx[wave][ixj] = ti;
                    }
                }
            }
            __syncthreads();
        }
    }

    // compositing: 4 consecutive sorted samples per lane
    int base = 4 * lane;
    float zq[4], alq[4], tq[4];
#pragma unroll
    for (int q = 0; q < 4; ++q) {
        int m = base + q;
        float z = s_key[wave][m];
        float d = (m < 127) ? (s_key[wave][m+1] - z) : 1e10f;
        int id = s_idx[wave][m];
        float sg = (id < 64) ? sigma_c[ray*64 + id] : sigma_f[ray*64 + id - 64];
        sg = fmaxf(sg, 0.0f);
        float al = 1.0f - __expf(-sg * d);
        zq[q] = z; alq[q] = al; tq[q] = 1.0f - al + 1e-10f;
    }
    float s = tq[0]*tq[1]*tq[2]*tq[3];
    for (int off = 1; off < 32; off <<= 1) { float v = __shfl_up(s, off, 32); if (lane >= off) s *= v; }
    float tr = __shfl_up(s, 1, 32); if (lane == 0) tr = 1.0f;

    float rr = 0, gg = 0, bb = 0, dep = 0, msk = 0;
#pragma unroll
    for (int q = 0; q < 4; ++q) {
        int m = base + q;
        float w = alq[q] * tr;
        int id = s_idx[wave][m];
        const float* rp = (id < 64) ? (rgb_c + (ray*64 + id)*3)
                                    : (rgb_f + (ray*64 + id - 64)*3);
        rr += w * rp[0]; gg += w * rp[1]; bb += w * rp[2];
        dep += w * zq[q]; msk += w;
        tr *= tq[q];
    }
    for (int off = 16; off; off >>= 1) {
        rr  += __shfl_xor(rr,  off, 32);
        gg  += __shfl_xor(gg,  off, 32);
        bb  += __shfl_xor(bb,  off, 32);
        dep += __shfl_xor(dep, off, 32);
        msk += __shfl_xor(msk, off, 32);
    }
    if (lane == 0) {
        float dx = rd[ray*3+0], dy = rd[ray*3+1], dz = rd[ray*3+2];
        const float* T = tform + (ray >> 12) * 16;          // ray / (64*64)
        float vz = dx*T[2] + dy*T[6] + dz*T[10];            // (R^T rd)_z
        out[ray*3+0] = rr; out[ray*3+1] = gg; out[ray*3+2] = bb;
        out[RAYS*3 + ray] = -vz * dep;
        out[RAYS*4 + ray] = msk;
    }
}

// ---------------------------------------------------------------------------
extern "C" void kernel_launch(void* const* d_in, const int* in_sizes, int n_in,
                              void* d_out, int out_size, void* d_ws, size_t ws_size,
                              hipStream_t stream)
{
    const float* org   = (const float*)d_in[0];
    const float* dir   = (const float*)d_in[1];
    const float* tform = (const float*)d_in[2];
    const float* noise = (const float*)d_in[3];
    const float* usamp = (const float*)d_in[4];
    const float* W1p   = (const float*)d_in[5];
    const float* b1p   = (const float*)d_in[6];
    const float* W2p   = (const float*)d_in[7];
    const float* b2p   = (const float*)d_in[8];
    (void)in_sizes; (void)n_in; (void)out_size; (void)ws_size;

    char* ws = (char*)d_ws;
    size_t o = 0;
    float* rd      = (float*)(ws + o); o += (size_t)RAYS*3*4;
    float* pts     = (float*)(ws + o); o += (size_t)NPTS*3*4;   // shared coarse/fine
    float* sigma_c = (float*)(ws + o); o += (size_t)NPTS*4;
    float* rgb_c   = (float*)(ws + o); o += (size_t)NPTS*3*4;
    float* sigma_f = (float*)(ws + o); o += (size_t)NPTS*4;
    float* rgb_f   = (float*)(ws + o); o += (size_t)NPTS*3*4;
    float* zsamp   = (float*)(ws + o); o += (size_t)NPTS*4;

    nerf_setup <<<NPTS/256, 256, 0, stream>>>(org, dir, noise, rd, pts);
    nerf_mlp   <<<NPTS/128, 256, 0, stream>>>(pts, W1p, b1p, W2p, b2p, sigma_c, rgb_c);
    nerf_sample<<<RAYS/8,   256, 0, stream>>>(noise, usamp, sigma_c, org, rd, zsamp, pts);
    nerf_mlp   <<<NPTS/128, 256, 0, stream>>>(pts, W1p, b1p, W2p, b2p, sigma_f, rgb_f);
    nerf_final <<<RAYS/8,   256, 0, stream>>>(noise, zsamp, sigma_c, rgb_c, sigma_f, rgb_f,
                                              rd, tform, (float*)d_out);
}
